// GATNet_45174466019798
// MI455X (gfx1250) — matrix operand order repaired and checked by reading, not compiled, not run on hardware
//
#include <hip/hip_runtime.h>
#include <hip/hip_bf16.h>

typedef __attribute__((ext_vector_type(16))) __bf16        v16bf;
typedef __attribute__((ext_vector_type(8)))  float          v8f;
typedef __attribute__((ext_vector_type(8)))  unsigned short v8u;
typedef __attribute__((ext_vector_type(16))) unsigned short v16u;

__device__ __forceinline__ float bf2f(unsigned short u) {
    return __uint_as_float(((unsigned)u) << 16);
}
__device__ __forceinline__ unsigned short f2bf(float f) {
    unsigned u = __float_as_uint(f);
    u += 0x7FFFu + ((u >> 16) & 1u);   // round-to-nearest-even
    return (unsigned short)(u >> 16);
}
// order-preserving float <-> uint (for atomicMax on floats incl. negatives)
__device__ __forceinline__ unsigned fenc(float f) {
    unsigned u = __float_as_uint(f);
    return (u >> 31) ? ~u : (u | 0x80000000u);
}
__device__ __forceinline__ float fdec(unsigned u) {
    return __uint_as_float((u & 0x80000000u) ? (u ^ 0x80000000u) : ~u);
}

// ---------------------------------------------------------------- utilities
__global__ void k_zero(unsigned* __restrict__ p, long long n) {
    long long i = (long long)blockIdx.x * blockDim.x + threadIdx.x;
    long long s = (long long)gridDim.x * blockDim.x;
    for (; i < n; i += s) p[i] = 0u;
}

__global__ void k_prep_x(const float* __restrict__ x, unsigned short* __restrict__ Xbf,
                         int N, int FIN, int KP) {
    int idx = blockIdx.x * blockDim.x + threadIdx.x;
    if (idx >= N * KP) return;
    int n = idx / KP, k = idx - n * KP;
    Xbf[idx] = (k < FIN) ? f2bf(x[(size_t)n * FIN + k]) : (unsigned short)0;
}

// W [K, Ncols] f32 -> WT [Ncols, KP] bf16 (zero-padded K)
__global__ void k_prep_wT(const float* __restrict__ W, unsigned short* __restrict__ WT,
                          int K, int KP, int Ncols) {
    int idx = blockIdx.x * blockDim.x + threadIdx.x;
    if (idx >= Ncols * KP) return;
    int c = idx / KP, k = idx - c * KP;
    WT[idx] = (k < K) ? f2bf(W[(size_t)k * Ncols + c]) : (unsigned short)0;
}

// -------------------------------------------------------- WMMA bf16 GEMM
// D[M,Ncols](bf16) = A[M,Ks](bf16, row-major) * BT[Ncols,Ks](bf16)^T
// One wave computes NT adjacent 16x16 tiles in the N dimension, sharing the
// A fragment across NT wmma issues per K-step (A traffic / NT, denser wmma).
// M%16==0, (Ncols/16)%NT==0, Ks%32==0.
template <int NT>
__global__ void k_wmma_gemm(const unsigned short* __restrict__ A,
                            const unsigned short* __restrict__ BT,
                            unsigned short* __restrict__ D,
                            int M, int Ncols, int Ks) {
    int wid  = (int)((blockIdx.x * blockDim.x + threadIdx.x) >> 5);
    int lane = threadIdx.x & 31;
    int ngrp = (Ncols >> 4) / NT;
    int mt   = wid / ngrp;
    int ng   = wid - mt * ngrp;
    if (mt >= (M >> 4)) return;           // wave-uniform exit: EXEC all-1 below
    int half = lane >> 4;
    int l15  = lane & 15;
    const unsigned short* arow = A + (size_t)(mt * 16 + l15) * Ks;
    const unsigned short* bcol[NT];
#pragma unroll
    for (int i = 0; i < NT; ++i)
        bcol[i] = BT + (size_t)((ng * NT + i) * 16 + l15) * Ks;

    v8f acc[NT] = {};
    for (int k0 = 0; k0 < Ks; k0 += 32) {
        // A 16x32 bf16 fragment layout (ISA 7.12.2):
        //   half==0: K = {k0+0..7, k0+16..23}; half==1: K = {k0+8..15, k0+24..31}
        v8u a0 = *(const v8u*)(arow + k0 + half * 8);
        v8u a1 = *(const v8u*)(arow + k0 + 16 + half * 8);
        v16u au = __builtin_shufflevector(a0, a1,
                    0, 1, 2, 3, 4, 5, 6, 7, 8, 9, 10, 11, 12, 13, 14, 15);
        v16bf av = __builtin_bit_cast(v16bf, au);
#pragma unroll
        for (int i = 0; i < NT; ++i) {
            // B 32x16 bf16 fragment: col = lane&15, K = {half*8..+7, 16+half*8..+7}
            v8u b0 = *(const v8u*)(bcol[i] + k0 + half * 8);
            v8u b1 = *(const v8u*)(bcol[i] + k0 + 16 + half * 8);
            v16u bu = __builtin_shufflevector(b0, b1,
                        0, 1, 2, 3, 4, 5, 6, 7, 8, 9, 10, 11, 12, 13, 14, 15);
            acc[i] = __builtin_amdgcn_wmma_f32_16x16x32_bf16(
                         false, av,
                         false, __builtin_bit_cast(v16bf, bu),
                         (short)0, acc[i], false, false);
        }
    }
    // C/D layout: vgpr r, lane -> (M = r + 8*half, N = lane&15)
#pragma unroll
    for (int i = 0; i < NT; ++i) {
        unsigned short* drow = D + (size_t)(mt * 16 + half * 8) * Ncols
                                 + (ng * NT + i) * 16 + l15;
#pragma unroll
        for (int r = 0; r < 8; ++r)
            drow[(size_t)r * Ncols] = f2bf(acc[i][r]);
    }
}

// ------------------------------------------------- GAT attention pipeline
__global__ void k_scores(const unsigned short* __restrict__ Hbf,
                         const float* __restrict__ a_s, const float* __restrict__ a_d,
                         float* __restrict__ ss, float* __restrict__ sd,
                         int N, int heads, int C) {
    int idx = blockIdx.x * blockDim.x + threadIdx.x;
    if (idx >= N * heads) return;
    int n = idx / heads, h = idx - n * heads;
    const unsigned short* row = Hbf + (size_t)n * heads * C + (size_t)h * C;
    float s = 0.f, d = 0.f;
    for (int c = 0; c < C; ++c) {
        float v = bf2f(row[c]);
        s += v * a_s[h * C + c];
        d += v * a_d[h * C + c];
    }
    ss[idx] = s; sd[idx] = d;
}

__global__ void k_edge_max(const int* __restrict__ srcs, const int* __restrict__ dsts,
                           int E, int Nn, int heads,
                           const float* __restrict__ ss, const float* __restrict__ sd,
                           unsigned* __restrict__ mu) {
    int idx = blockIdx.x * blockDim.x + threadIdx.x;
    if (idx >= (E + Nn) * heads) return;
    int e = idx / heads, h = idx - e * heads;
    int s, d;
    if (e < E) { s = srcs[e]; d = dsts[e]; } else { s = d = e - E; }   // self loop
    float v = ss[s * heads + h] + sd[d * heads + h];
    v = (v > 0.f) ? v : 0.2f * v;                                      // LeakyReLU
    atomicMax(&mu[d * heads + h], fenc(v));
}

__global__ void k_edge_sum(const int* __restrict__ srcs, const int* __restrict__ dsts,
                           int E, int Nn, int heads,
                           const float* __restrict__ ss, const float* __restrict__ sd,
                           const unsigned* __restrict__ mu, float* __restrict__ z) {
    int idx = blockIdx.x * blockDim.x + threadIdx.x;
    if (idx >= (E + Nn) * heads) return;
    int e = idx / heads, h = idx - e * heads;
    int s, d;
    if (e < E) { s = srcs[e]; d = dsts[e]; } else { s = d = e - E; }
    float v = ss[s * heads + h] + sd[d * heads + h];
    v = (v > 0.f) ? v : 0.2f * v;
    float p = __expf(v - fdec(mu[d * heads + h]));
    unsafeAtomicAdd(&z[d * heads + h], p);
}

// one wave per (edge, head): lanes stripe the C channels
__global__ void k_edge_scatter(const int* __restrict__ srcs, const int* __restrict__ dsts,
                               int E, int Nn, int heads, int C,
                               const float* __restrict__ ss, const float* __restrict__ sd,
                               const unsigned* __restrict__ mu, const float* __restrict__ z,
                               const unsigned short* __restrict__ Hbf,
                               float* __restrict__ AGG) {
    int wid  = (int)((blockIdx.x * blockDim.x + threadIdx.x) >> 5);
    int lane = threadIdx.x & 31;
    if (wid >= (E + Nn) * heads) return;
    int e = wid / heads, h = wid - e * heads;
    int s, d;
    if (e < E) { s = srcs[e]; d = dsts[e]; } else { s = d = e - E; }
    const unsigned short* hrow = Hbf + (size_t)s * heads * C + (size_t)h * C;
    __builtin_prefetch(hrow + lane * 4, 0, 0);   // start gather early (global_prefetch)
    float v = ss[s * heads + h] + sd[d * heads + h];
    v = (v > 0.f) ? v : 0.2f * v;
    float alpha = __expf(v - fdec(mu[d * heads + h])) / (z[d * heads + h] + 1e-16f);
    float* arow = AGG + (size_t)d * heads * C + (size_t)h * C;
    for (int c = lane; c < C; c += 32)
        unsafeAtomicAdd(arow + c, alpha * bf2f(hrow[c]));
}

// AGG1 + b1 -> ELU -> bf16 (becomes layer-2 input)
__global__ void k_finish1(const float* __restrict__ AGG, const float* __restrict__ bias,
                          unsigned short* __restrict__ Xbf, long long n, int HD) {
    long long idx = (long long)blockIdx.x * blockDim.x + threadIdx.x;
    if (idx >= n) return;
    int j = (int)(idx % HD);
    float v = AGG[idx] + bias[j];
    v = (v > 0.f) ? v : (__expf(v) - 1.f);
    Xbf[idx] = f2bf(v);
}

// AGG2 + b2 -> ReLU in place (f32)
__global__ void k_finish2(float* __restrict__ AGG, const float* __restrict__ bias,
                          long long n, int C) {
    long long idx = (long long)blockIdx.x * blockDim.x + threadIdx.x;
    if (idx >= n) return;
    float v = AGG[idx] + bias[(int)(idx % C)];
    AGG[idx] = (v > 0.f) ? v : 0.f;
}

// global max pool over batch assignment
__global__ void k_pool(const float* __restrict__ h2, const int* __restrict__ batch,
                       unsigned* __restrict__ gmax, long long n, int C) {
    long long idx = (long long)blockIdx.x * blockDim.x + threadIdx.x;
    if (idx >= n) return;
    int nn = (int)(idx / C), c = (int)(idx - (long long)nn * C);
    atomicMax(&gmax[batch[nn] * C + c], fenc(h2[idx]));
}

// relu(g @ wg + bg)
__global__ void k_gmlp(const unsigned* __restrict__ gmax, const float* __restrict__ wg,
                       const float* __restrict__ bg, float* __restrict__ g2, int B) {
    int idx = blockIdx.x * blockDim.x + threadIdx.x;
    if (idx >= B * 144) return;
    int b = idx / 144, j = idx - b * 144;
    float acc = bg[j];
    for (int k = 0; k < 144; ++k)
        acc += fdec(gmax[b * 144 + k]) * wg[k * 144 + j];
    g2[idx] = (acc > 0.f) ? acc : 0.f;
}

// 3x3 SAME conv on [B,8,36,2] + bias + relu + 2x2 maxpool -> pfeat[B,144]
__global__ void k_conv(const float* __restrict__ t, const float* __restrict__ w,
                       const float* __restrict__ cb, float* __restrict__ pfeat, int B) {
    int idx = blockIdx.x * blockDim.x + threadIdx.x;
    if (idx >= B * 8 * 18) return;
    int b = idx / (8 * 18); int r = idx - b * 8 * 18;
    int o = r / 18, hh = r - o * 18;
    float best = 0.f;                      // values post-relu are >= 0
    for (int dy = 0; dy < 2; ++dy) {
        int y = 2 * hh + dy;
        for (int x = 0; x < 2; ++x) {
            float acc = cb[o];
            for (int i = 0; i < 8; ++i)
                for (int kh = 0; kh < 3; ++kh) {
                    int iy = y + kh - 1; if (iy < 0 || iy >= 36) continue;
                    for (int kw = 0; kw < 3; ++kw) {
                        int ix = x + kw - 1; if (ix < 0 || ix >= 2) continue;
                        acc += t[((b * 8 + i) * 36 + iy) * 2 + ix] *
                               w[((o * 8 + i) * 3 + kh) * 3 + kw];
                    }
                }
            acc = (acc > 0.f) ? acc : 0.f;
            best = fmaxf(best, acc);
        }
    }
    pfeat[b * 144 + o * 18 + hh] = best;
}

__global__ void k_fc1(const float* __restrict__ g2, const float* __restrict__ pf,
                      const float* __restrict__ w, const float* __restrict__ bias,
                      float* __restrict__ f1, int B) {
    int idx = blockIdx.x * blockDim.x + threadIdx.x;
    if (idx >= B * 144) return;
    int b = idx / 144, j = idx - b * 144;
    float acc = bias[j];
    for (int k = 0; k < 144; ++k) acc += g2[b * 144 + k] * w[k * 144 + j];
    for (int k = 0; k < 144; ++k) acc += pf[b * 144 + k] * w[(144 + k) * 144 + j];
    f1[idx] = (acc > 0.f) ? acc : 0.01f * acc;
}

__global__ void k_fc2(const float* __restrict__ f1, const float* __restrict__ w,
                      const float* __restrict__ bias, float* __restrict__ f2, int B) {
    int idx = blockIdx.x * blockDim.x + threadIdx.x;
    if (idx >= B * 72) return;
    int b = idx / 72, j = idx - b * 72;
    float acc = bias[j];
    for (int k = 0; k < 144; ++k) acc += f1[b * 144 + k] * w[k * 72 + j];
    f2[idx] = (acc > 0.f) ? acc : 0.01f * acc;
}

__global__ void k_fc3(const float* __restrict__ f2, const float* __restrict__ w,
                      const float* __restrict__ bias, float* __restrict__ out, int B) {
    int b = blockIdx.x * blockDim.x + threadIdx.x;
    if (b >= B) return;
    float acc = bias[0];
    for (int k = 0; k < 72; ++k) acc += f2[b * 72 + k] * w[k];
    out[b] = acc;
}

// ----------------------------------------------------------------- launch
extern "C" void kernel_launch(void* const* d_in, const int* in_sizes, int n_in,
                              void* d_out, int out_size, void* d_ws, size_t ws_size,
                              hipStream_t stream) {
    const float* x   = (const float*)d_in[0];
    const int*   ei  = (const int*)d_in[1];
    const int*   bat = (const int*)d_in[2];
    const float* tg  = (const float*)d_in[3];
    const float* w1  = (const float*)d_in[4];
    const float* as1 = (const float*)d_in[5];
    const float* ad1 = (const float*)d_in[6];
    const float* b1  = (const float*)d_in[7];
    const float* w2  = (const float*)d_in[8];
    const float* as2 = (const float*)d_in[9];
    const float* ad2 = (const float*)d_in[10];
    const float* b2  = (const float*)d_in[11];
    const float* wg  = (const float*)d_in[12];
    const float* bg  = (const float*)d_in[13];
    const float* cw  = (const float*)d_in[14];
    const float* cb  = (const float*)d_in[15];
    const float* wf1 = (const float*)d_in[16];
    const float* bfc1= (const float*)d_in[17];
    const float* wf2 = (const float*)d_in[18];
    const float* bfc2= (const float*)d_in[19];
    const float* wf3 = (const float*)d_in[20];
    const float* bfc3= (const float*)d_in[21];
    float* out = (float*)d_out;

    const int FIN = 78, KP = 96, H1 = 10, C1 = 144, HD1 = 1440, C2 = 144;
    const int N = in_sizes[0] / FIN;          // 50000 (divisible by 16)
    const int E = in_sizes[1] / 2;            // 100000
    const int B = in_sizes[3] / (8 * 36 * 2); // 512
    const int Etot = E + N;

    // ---- workspace carve-up (256B aligned)
    size_t off = 0; char* base = (char*)d_ws;
    auto alloc = [&](size_t bytes) -> void* {
        void* p = base + off;
        off = (off + bytes + 255) & ~(size_t)255;
        return p;
    };
    unsigned short* Xbf  = (unsigned short*)alloc((size_t)N * KP * 2);
    unsigned short* W1T  = (unsigned short*)alloc((size_t)HD1 * KP * 2);
    unsigned short* H1bf = (unsigned short*)alloc((size_t)N * HD1 * 2); // reused as X2bf
    float*          AGG1 = (float*)alloc((size_t)N * HD1 * 4);
    float*          ss1  = (float*)alloc((size_t)N * H1 * 4);
    float*          sd1  = (float*)alloc((size_t)N * H1 * 4);
    unsigned*       m1u  = (unsigned*)alloc((size_t)N * H1 * 4);
    float*          z1   = (float*)alloc((size_t)N * H1 * 4);
    unsigned short* W2T  = (unsigned short*)alloc((size_t)C2 * HD1 * 2);
    unsigned short* H2bf = (unsigned short*)alloc((size_t)N * C2 * 2);
    float*          ss2  = (float*)alloc((size_t)N * 4);
    float*          sd2  = (float*)alloc((size_t)N * 4);
    unsigned*       m2u  = (unsigned*)alloc((size_t)N * 4);
    float*          z2   = (float*)alloc((size_t)N * 4);
    float*          AGG2 = (float*)alloc((size_t)N * C2 * 4);   // becomes h2
    unsigned*       gmax = (unsigned*)alloc((size_t)B * 144 * 4);
    float*          g2   = (float*)alloc((size_t)B * 144 * 4);
    float*          pfeat= (float*)alloc((size_t)B * 144 * 4);
    float*          f1   = (float*)alloc((size_t)B * 144 * 4);
    float*          f2b  = (float*)alloc((size_t)B * 72 * 4);
    (void)ws_size; (void)n_in; (void)out_size;

    const int T = 256;
    auto G = [](long long n) { return dim3((unsigned)((n + 255) / 256)); };

    // ---- zero accumulators / reduction state (every call)
    k_zero<<<4096, T, 0, stream>>>((unsigned*)AGG1, (long long)N * HD1);
    k_zero<<<256,  T, 0, stream>>>(m1u, (long long)N * H1);
    k_zero<<<256,  T, 0, stream>>>((unsigned*)z1, (long long)N * H1);
    k_zero<<<1024, T, 0, stream>>>((unsigned*)AGG2, (long long)N * C2);
    k_zero<<<64,   T, 0, stream>>>(m2u, N);
    k_zero<<<64,   T, 0, stream>>>((unsigned*)z2, N);
    k_zero<<<64,   T, 0, stream>>>(gmax, (long long)B * 144);

    // ---- bf16 prep
    k_prep_x <<<G((long long)N * KP), T, 0, stream>>>(x, Xbf, N, FIN, KP);
    k_prep_wT<<<G((long long)HD1 * KP), T, 0, stream>>>(w1, W1T, FIN, KP, HD1);
    k_prep_wT<<<G((long long)C2 * HD1), T, 0, stream>>>(w2, W2T, HD1, HD1, C2);

    // ---- layer 1: GEMM (WMMA, 6 N-tiles/wave) + attention + scatter + ELU
    {
        long long waves = (long long)(N / 16) * ((HD1 / 16) / 6);
        k_wmma_gemm<6><<<G(waves * 32), T, 0, stream>>>(Xbf, W1T, H1bf, N, HD1, KP);
    }
    k_scores<<<G((long long)N * H1), T, 0, stream>>>(H1bf, as1, ad1, ss1, sd1, N, H1, C1);
    k_edge_max<<<G((long long)Etot * H1), T, 0, stream>>>(ei, ei + E, E, N, H1, ss1, sd1, m1u);
    k_edge_sum<<<G((long long)Etot * H1), T, 0, stream>>>(ei, ei + E, E, N, H1, ss1, sd1, m1u, z1);
    k_edge_scatter<<<G((long long)Etot * H1 * 32), T, 0, stream>>>(
        ei, ei + E, E, N, H1, C1, ss1, sd1, m1u, z1, H1bf, AGG1);
    k_finish1<<<G((long long)N * HD1), T, 0, stream>>>(AGG1, b1, H1bf, (long long)N * HD1, HD1);

    // ---- layer 2: GEMM (WMMA, 3 N-tiles/wave) + attention + scatter + ReLU
    {
        long long waves = (long long)(N / 16) * ((C2 / 16) / 3);
        k_wmma_gemm<3><<<G(waves * 32), T, 0, stream>>>(H1bf, W2T, H2bf, N, C2, HD1);
    }
    k_scores<<<G((long long)N), T, 0, stream>>>(H2bf, as2, ad2, ss2, sd2, N, 1, C2);
    k_edge_max<<<G((long long)Etot), T, 0, stream>>>(ei, ei + E, E, N, 1, ss2, sd2, m2u);
    k_edge_sum<<<G((long long)Etot), T, 0, stream>>>(ei, ei + E, E, N, 1, ss2, sd2, m2u, z2);
    k_edge_scatter<<<G((long long)Etot * 32), T, 0, stream>>>(
        ei, ei + E, E, N, 1, C2, ss2, sd2, m2u, z2, H2bf, AGG2);
    k_finish2<<<G((long long)N * C2), T, 0, stream>>>(AGG2, b2, (long long)N * C2, C2);

    // ---- pooling + head
    k_pool<<<G((long long)N * C2), T, 0, stream>>>(AGG2, bat, gmax, (long long)N * C2, C2);
    k_gmlp<<<G((long long)B * 144), T, 0, stream>>>(gmax, wg, bg, g2, B);
    k_conv<<<G((long long)B * 8 * 18), T, 0, stream>>>(tg, cw, cb, pfeat, B);
    k_fc1<<<G((long long)B * 144), T, 0, stream>>>(g2, pfeat, wf1, bfc1, f1, B);
    k_fc2<<<G((long long)B * 72), T, 0, stream>>>(f1, wf2, bfc2, f2b, B);
    k_fc3<<<G((long long)B), T, 0, stream>>>(f2b, wf3, bfc3, out, B);
}
